// Attention_41669772706202
// MI455X (gfx1250) — matrix operand order repaired and checked by reading, not compiled, not run on hardware
//
#include <hip/hip_runtime.h>

typedef _Float16 f16;
typedef __attribute__((ext_vector_type(16))) _Float16 v16h;
typedef __attribute__((ext_vector_type(8)))  _Float16 v8h;
typedef __attribute__((ext_vector_type(8)))  float    v8f;

#define B_   2
#define S_   2048
#define E_   1024
#define H_   16
#define D_   64
#define BH_  32
#define M_   4096            // B*S
static constexpr float ATTN_SCALE = 0.125f;  // 64^-0.5

// ---------------------------------------------------------------------------
// CDNA5 async copy: 16 bytes per lane, global -> LDS, tracked by ASYNCcnt.
// LDS byte offset = low 32 bits of the generic pointer (AS3->generic cast
// places the LDS offset in the low dword per the aperture mapping).
// ---------------------------------------------------------------------------
__device__ __forceinline__ void async_copy16(const f16* g, f16* l) {
  unsigned loff = (unsigned)(size_t)l;
  unsigned long long ga = (unsigned long long)(size_t)g;
  asm volatile("global_load_async_to_lds_b128 %0, %1, off"
               :: "v"(loff), "v"(ga) : "memory");
}
__device__ __forceinline__ void wait_async0() {
  asm volatile("s_wait_asynccnt 0x0" ::: "memory");
}

// ---------------------------------------------------------------------------
// WMMA fragment builders (wave32; this file's contiguous K order, same for A and B)
// A (16x32 f16): lane<16 -> row M=lane, K {k0..k0+15};
//                lane>=16 -> row M=lane-16, K {k0+16..k0+31}.
// B (32x16 f16): lane holds column N = n0+(lane&15); 16 contiguous K values
//                (upper half-wave takes K = k0+16..k0+31).
// ---------------------------------------------------------------------------
__device__ __forceinline__ v16h a_frag_rowmajor(const f16* base, int ld,
                                                int m0, int k0) {
  const int l = threadIdx.x & 31;
  const f16* p = base + (size_t)(m0 + (l & 15)) * ld + k0 + ((l >> 4) << 4);
  v8h lo = *(const v8h*)p;
  v8h hi = *(const v8h*)(p + 8);
  return __builtin_shufflevector(lo, hi, 0,1,2,3,4,5,6,7,8,9,10,11,12,13,14,15);
}
// tile: LDS, row-major [ncols][32] (one 32-long K strip per column)
__device__ __forceinline__ v16h b_frag_lds(const f16* tile, int n0) {
  const int l = threadIdx.x & 31;
  return *(const v16h*)(tile + (n0 + (l & 15)) * 32 + ((l >> 4) << 4));
}
// K tile in LDS is [32 keys][64 d]; B-frag column = key, K-dim = d chunk.
__device__ __forceinline__ v16h b_frag_lds_k(const f16* tile, int n0, int kd) {
  const int l = threadIdx.x & 31;
  return *(const v16h*)(tile + (n0 + (l & 15)) * 64 + kd + ((l >> 4) << 4));
}
// tile: LDS, row-major [nrows][32]; A fragment with K = 0..31 of the tile
__device__ __forceinline__ v16h a_frag_lds(const f16* tile, int m0) {
  const int l = threadIdx.x & 31;
  const f16* p = tile + (m0 + (l & 15)) * 32 + ((l >> 4) << 4);
  v8h lo = *(const v8h*)p;
  v8h hi = *(const v8h*)(p + 8);
  return __builtin_shufflevector(lo, hi, 0,1,2,3,4,5,6,7,8,9,10,11,12,13,14,15);
}

#define WMMA_F16(a, b, c) \
  __builtin_amdgcn_wmma_f32_16x16x32_f16(false, (a), false, (b), (short)0, (c), false, false)

// ---------------------------------------------------------------------------
// fp32 -> f16 cast
// ---------------------------------------------------------------------------
__global__ void cast_f32_to_f16(const float* __restrict__ src,
                                f16* __restrict__ dst, int n) {
  int i = blockIdx.x * blockDim.x + threadIdx.x;
  const int stride = gridDim.x * blockDim.x;
  for (; i < n; i += stride) dst[i] = (f16)src[i];
}

// ---------------------------------------------------------------------------
// GEMM: y[m,n] = (sum_k A[m,k] * W[n,k] + bias[n]) * scale
//   mode 0: f16 out, split heads  -> out16[((b*H+h)*S + s)*D + d]
//   mode 1: f16 out, transposed   -> out16[((b*H+h)*D + d)*S + s]   (for V)
//   mode 2: f32 out, plain        -> out32[m*N + n]
// Block: 256 thr = 8 waves; tile 64(M) x 128(N); wave tile 16x64.
// Double-buffered LDS staging filled by async global->LDS copies.
// ---------------------------------------------------------------------------
__global__ __launch_bounds__(256)
void gemm_wmma_kernel(const f16* __restrict__ A, const f16* __restrict__ W,
                      const float* __restrict__ bias,
                      f16* __restrict__ out16, float* __restrict__ out32,
                      int M, int N, int K, float scale, int mode) {
  __shared__ f16 Abuf[2][64 * 32];    // [row][k]   4 KB each
  __shared__ f16 Bbuf[2][128 * 32];   // [col][k]   8 KB each
  const int t    = threadIdx.x;
  const int wave = t >> 5;
  const int l    = t & 31;
  const int mblk = blockIdx.y * 64;
  const int nblk = blockIdx.x * 128;
  const int wm   = (wave & 3) * 16;
  const int wn   = (wave >> 2) * 64;

  // staging assignments: A = 256 16B chunks (1/thread), B = 512 (2/thread)
  const int arow = t >> 2,          ach = (t & 3) * 8;
  const int bc0  = t,               bc1 = t + 256;
  const int bcol0 = bc0 >> 2,       bch0 = (bc0 & 3) * 8;
  const int bcol1 = bc1 >> 2,       bch1 = (bc1 & 3) * 8;

  auto stage = [&](int buf, int k) {
    async_copy16(A + (size_t)(mblk + arow) * K + k + ach,
                 &Abuf[buf][arow * 32 + ach]);
    async_copy16(W + (size_t)(nblk + bcol0) * K + k + bch0,
                 &Bbuf[buf][bcol0 * 32 + bch0]);
    async_copy16(W + (size_t)(nblk + bcol1) * K + k + bch1,
                 &Bbuf[buf][bcol1 * 32 + bch1]);
  };

  stage(0, 0);
  wait_async0();
  __syncthreads();

  v8f c0 = {}, c1 = {}, c2 = {}, c3 = {};
  const int NK = K / 32;
  for (int kt = 0; kt < NK; ++kt) {
    const int cur = kt & 1;
    if (kt + 1 < NK) stage(1 - cur, (kt + 1) * 32);   // overlap with compute
    v16h a  = a_frag_lds(Abuf[cur], wm);
    v16h b0 = b_frag_lds(Bbuf[cur], wn +  0);
    v16h b1 = b_frag_lds(Bbuf[cur], wn + 16);
    v16h b2 = b_frag_lds(Bbuf[cur], wn + 32);
    v16h b3 = b_frag_lds(Bbuf[cur], wn + 48);
    c0 = WMMA_F16(a, b0, c0);
    c1 = WMMA_F16(a, b1, c1);
    c2 = WMMA_F16(a, b2, c2);
    c3 = WMMA_F16(a, b3, c3);
    wait_async0();
    __syncthreads();
  }

  // C layout: element (M = r + 8*(lane>=16), N = lane&15) lives in c[r]
  const int mrow  = mblk + wm + ((l >> 4) << 3);
  const int ncol0 = nblk + wn + (l & 15);
#pragma unroll
  for (int r = 0; r < 8; ++r) {
    const int m  = mrow + r;
    const int bb = m >> 11;           // / S_
    const int s  = m & (S_ - 1);
    float v[4] = { c0[r], c1[r], c2[r], c3[r] };
#pragma unroll
    for (int tt = 0; tt < 4; ++tt) {
      const int n = ncol0 + tt * 16;
      const float y = (v[tt] + bias[n]) * scale;
      if (mode == 2) {
        out32[(size_t)m * N + n] = y;
      } else {
        const int h = n >> 6, d = n & 63;
        if (mode == 0)
          out16[(((size_t)(bb * H_ + h)) * S_ + s) * D_ + d] = (f16)y;
        else
          out16[(((size_t)(bb * H_ + h)) * D_ + d) * S_ + s] = (f16)y;
      }
    }
  }
}

// ---------------------------------------------------------------------------
// Flash attention: 4 waves/block share one head (bh); each wave owns a
// 16-query tile. 32-key K/V tiles are staged once per block into
// double-buffered LDS via async copies and shared by all 4 waves.
// Q,K: [BH,S,D] f16 (Q pre-scaled); V: [BH,D,S] f16; out: [B,S,E] f16.
// ---------------------------------------------------------------------------
__global__ __launch_bounds__(128)
void attn_wmma_kernel(const f16* __restrict__ Q, const f16* __restrict__ Kt,
                      const f16* __restrict__ Vt, f16* __restrict__ Aout) {
  __shared__ f16 Kbuf[2][32 * 64];   // [key][d]  4 KB each
  __shared__ f16 Vbuf[2][64 * 32];   // [d][s]    4 KB each
  __shared__ f16 plds[4][16 * 32];   // per-wave P transpose scratch
  const int t    = threadIdx.x;
  const int wave = t >> 5;
  const int l    = t & 31;
  const int bh   = blockIdx.x >> 5;                       // 0..31
  const int q0   = ((blockIdx.x & 31) * 4 + wave) * 16;   // query tile base

  const f16* Qb = Q  + (size_t)bh * S_ * D_;
  const f16* Kb = Kt + (size_t)bh * S_ * D_;
  const f16* Vb = Vt + (size_t)bh * D_ * S_;
  f16* pt = plds[wave];

  // staging: K tile 256 chunks (rows of 8x16B), V tile 256 chunks (rows of 4)
  const int kc0 = t, kc1 = t + 128;
  const int vc0 = t, vc1 = t + 128;
  auto stage = [&](int buf, int s0) {
    async_copy16(Kb + (size_t)(s0 + (kc0 >> 3)) * D_ + (kc0 & 7) * 8,
                 &Kbuf[buf][(kc0 >> 3) * 64 + (kc0 & 7) * 8]);
    async_copy16(Kb + (size_t)(s0 + (kc1 >> 3)) * D_ + (kc1 & 7) * 8,
                 &Kbuf[buf][(kc1 >> 3) * 64 + (kc1 & 7) * 8]);
    async_copy16(Vb + (size_t)(vc0 >> 2) * S_ + s0 + (vc0 & 3) * 8,
                 &Vbuf[buf][(vc0 >> 2) * 32 + (vc0 & 3) * 8]);
    async_copy16(Vb + (size_t)(vc1 >> 2) * S_ + s0 + (vc1 & 3) * 8,
                 &Vbuf[buf][(vc1 >> 2) * 32 + (vc1 & 3) * 8]);
  };

  const v16h qa0 = a_frag_rowmajor(Qb, D_, q0, 0);
  const v16h qa1 = a_frag_rowmajor(Qb, D_, q0, 32);

  v8f o0 = {}, o1 = {}, o2 = {}, o3 = {};
  float mrun[8], lrun[8];
#pragma unroll
  for (int r = 0; r < 8; ++r) { mrun[r] = -1e30f; lrun[r] = 0.0f; }

  stage(0, 0);
  wait_async0();
  __syncthreads();

  for (int s0 = 0; s0 < S_; s0 += 32) {
    const int cur = (s0 >> 5) & 1;
    if (s0 + 32 < S_) stage(1 - cur, s0 + 32);          // overlap with compute

    // scores: S(16x32) = Q(16x64) x K^T from LDS; K-frag columns = key rows
    v16h kb00 = b_frag_lds_k(Kbuf[cur],  0,  0);
    v16h kb01 = b_frag_lds_k(Kbuf[cur],  0, 32);
    v16h kb10 = b_frag_lds_k(Kbuf[cur], 16,  0);
    v16h kb11 = b_frag_lds_k(Kbuf[cur], 16, 32);
    v8f sc0 = {}, sc1 = {};
    sc0 = WMMA_F16(qa0, kb00, sc0);
    sc0 = WMMA_F16(qa1, kb01, sc0);
    sc1 = WMMA_F16(qa0, kb10, sc1);
    sc1 = WMMA_F16(qa1, kb11, sc1);

    // online softmax; row r lives across one half-wave's 16 lanes
#pragma unroll
    for (int r = 0; r < 8; ++r) {
      float mx = fmaxf(sc0[r], sc1[r]);
      mx = fmaxf(mx, __shfl_xor(mx, 1));
      mx = fmaxf(mx, __shfl_xor(mx, 2));
      mx = fmaxf(mx, __shfl_xor(mx, 4));
      mx = fmaxf(mx, __shfl_xor(mx, 8));
      const float mnew  = fmaxf(mrun[r], mx);
      const float alpha = __expf(mrun[r] - mnew);
      const float p0 = __expf(sc0[r] - mnew);
      const float p1 = __expf(sc1[r] - mnew);
      float ps = p0 + p1;
      ps += __shfl_xor(ps, 1);
      ps += __shfl_xor(ps, 2);
      ps += __shfl_xor(ps, 4);
      ps += __shfl_xor(ps, 8);
      lrun[r] = lrun[r] * alpha + ps;
      mrun[r] = mnew;
      o0[r] *= alpha; o1[r] *= alpha; o2[r] *= alpha; o3[r] *= alpha;
      const int Mr = r + ((l >> 4) << 3);
      const int n  = l & 15;
      pt[Mr * 32 + n]      = (f16)p0;
      pt[Mr * 32 + 16 + n] = (f16)p1;
    }
    asm volatile("s_wait_dscnt 0x0" ::: "memory");

    // O(16x64) += P(16x32) x V(32x64), all operands in LDS
    {
      v16h pa  = a_frag_lds(pt, 0);
      v16h vb0 = b_frag_lds(Vbuf[cur],  0);
      v16h vb1 = b_frag_lds(Vbuf[cur], 16);
      v16h vb2 = b_frag_lds(Vbuf[cur], 32);
      v16h vb3 = b_frag_lds(Vbuf[cur], 48);
      o0 = WMMA_F16(pa, vb0, o0);
      o1 = WMMA_F16(pa, vb1, o1);
      o2 = WMMA_F16(pa, vb2, o2);
      o3 = WMMA_F16(pa, vb3, o3);
    }
    wait_async0();
    __syncthreads();
  }

  // normalize, store merged heads: Aout[(b*S+s)*E + h*D + dtile*16 + n]
  const int mrow = (l >> 4) << 3;
  const int ncol = l & 15;
  const int bb = bh >> 4;
  const int h  = bh & (H_ - 1);
#pragma unroll
  for (int r = 0; r < 8; ++r) {
    const int s = q0 + mrow + r;
    const float inv = 1.0f / lrun[r];
    const size_t rowbase = ((size_t)(bb * S_ + s)) * E_ + h * D_;
    Aout[rowbase +  0 + ncol] = (f16)(o0[r] * inv);
    Aout[rowbase + 16 + ncol] = (f16)(o1[r] * inv);
    Aout[rowbase + 32 + ncol] = (f16)(o2[r] * inv);
    Aout[rowbase + 48 + ncol] = (f16)(o3[r] * inv);
  }
}

// ---------------------------------------------------------------------------
extern "C" void kernel_launch(void* const* d_in, const int* in_sizes, int n_in,
                              void* d_out, int out_size, void* d_ws, size_t ws_size,
                              hipStream_t stream) {
  (void)in_sizes; (void)n_in; (void)out_size; (void)ws_size;
  const float* x   = (const float*)d_in[0];
  const float* q_w = (const float*)d_in[1];
  const float* q_b = (const float*)d_in[2];
  const float* k_w = (const float*)d_in[3];
  const float* k_b = (const float*)d_in[4];
  const float* v_w = (const float*)d_in[5];
  const float* v_b = (const float*)d_in[6];
  const float* o_w = (const float*)d_in[7];
  const float* o_b = (const float*)d_in[8];
  float* out = (float*)d_out;

  const size_t ME = (size_t)M_ * E_;   // 4M elements
  const size_t EE = (size_t)E_ * E_;   // 1M elements
  f16* xh  = (f16*)d_ws;
  f16* qwh = xh  + ME;
  f16* kwh = qwh + EE;
  f16* vwh = kwh + EE;
  f16* owh = vwh + EE;
  f16* Qh  = owh + EE;     // [BH,S,D]
  f16* Kh  = Qh  + ME;     // [BH,S,D]
  f16* Vt  = Kh  + ME;     // [BH,D,S]
  f16* Ah  = Vt  + ME;     // [B,S,E]

  cast_f32_to_f16<<<512, 256, 0, stream>>>(x,   xh,  (int)ME);
  cast_f32_to_f16<<<256, 256, 0, stream>>>(q_w, qwh, (int)EE);
  cast_f32_to_f16<<<256, 256, 0, stream>>>(k_w, kwh, (int)EE);
  cast_f32_to_f16<<<256, 256, 0, stream>>>(v_w, vwh, (int)EE);
  cast_f32_to_f16<<<256, 256, 0, stream>>>(o_w, owh, (int)EE);

  dim3 gblk(E_ / 128, M_ / 64);  // (8, 64)
  gemm_wmma_kernel<<<gblk, 256, 0, stream>>>(xh, qwh, q_b, Qh, nullptr, M_, E_, E_, ATTN_SCALE, 0);
  gemm_wmma_kernel<<<gblk, 256, 0, stream>>>(xh, kwh, k_b, Kh, nullptr, M_, E_, E_, 1.0f, 0);
  gemm_wmma_kernel<<<gblk, 256, 0, stream>>>(xh, vwh, v_b, Vt, nullptr, M_, E_, E_, 1.0f, 1);

  attn_wmma_kernel<<<(BH_ * (S_ / 16)) / 4, 128, 0, stream>>>(Qh, Kh, Vt, Ah);

  gemm_wmma_kernel<<<gblk, 256, 0, stream>>>(Ah, owh, o_b, nullptr, out, M_, E_, E_, 1.0f, 2);
}
